// PatchedDeepseekV2Attention_6047313953504
// MI455X (gfx1250) — compile-verified, matrix operand order checked by reading, not run on hardware
//
#include <hip/hip_runtime.h>
#include <math.h>

// ---------------- problem constants ----------------
#define H_    32
#define DN_   128
#define DR_   64
#define DV_   128
#define KVL_  512
#define QL_   1536
#define HID_  4096
#define S_    2048
#define CFULL 576          // KVL + DR : q_full/k_full dim
#define QDIM  192          // DN + DR per head
#define SCALE_ 0.07216878364870323f   // (DN+DR)^-0.5

// ---------------- types ----------------
typedef __attribute__((ext_vector_type(16))) __bf16 bf16x16;
typedef __attribute__((ext_vector_type(8)))  float  floatx8;
typedef __attribute__((ext_vector_type(4))) unsigned tdm_g0_t;
typedef __attribute__((ext_vector_type(8))) int      tdm_g1_t;
typedef __attribute__((ext_vector_type(4))) int      tdm_g2_t;
typedef __attribute__((ext_vector_type(8))) int      tdm_g4_t;

union BFrag { bf16x16 v; unsigned u[8]; };
union AccF  { floatx8 v; float f[8]; };

// f32 -> bf16 bits (round-to-nearest-even)
__device__ __forceinline__ unsigned short f2b(float f) {
  union { float f; unsigned u; } c; c.f = f;
  unsigned r = c.u + 0x7FFFu + ((c.u >> 16) & 1u);
  return (unsigned short)(r >> 16);
}
__device__ __forceinline__ float b2f(unsigned short b) {
  union { unsigned u; float f; } c; c.u = ((unsigned)b) << 16;
  return c.f;
}

__device__ __forceinline__ floatx8 wmma_bf16(bf16x16 a, bf16x16 b, floatx8 c) {
  return __builtin_amdgcn_wmma_f32_16x16x32_bf16(false, a, false, b, (short)0, c,
                                                 false, false);
}

// A-fragment (16x32 bf16, row-major source, K contiguous within row).
// Per ISA 7.12.2: lanes 0-15 = row M, VGPR p holds K pairs
// {0,1},{2,3},{4,5},{6,7},{16,17},{18,19},{20,21},{22,23}; lanes 16-31: +8.
__device__ __forceinline__ bf16x16 load_frag_row(const unsigned short* rowbase) {
  BFrag f;
  const int half8 = (threadIdx.x & 16) ? 8 : 0;
#pragma unroll
  for (int p = 0; p < 8; ++p) {
    int k = ((p < 4) ? (2 * p) : (16 + 2 * (p - 4))) + half8;
    f.u[p] = *(const unsigned*)(rowbase + k);
  }
  return f.v;
}

// B-fragment from column-major-ish source: element (k, col) at base[k*ld]
// (base already offset by column). Used for P*V where ckv rows are keys.
__device__ __forceinline__ bf16x16 load_frag_col(const unsigned short* base, int ld) {
  BFrag f;
  const int half8 = (threadIdx.x & 16) ? 8 : 0;
#pragma unroll
  for (int p = 0; p < 8; ++p) {
    int k = ((p < 4) ? (2 * p) : (16 + 2 * (p - 4))) + half8;
    unsigned lo = base[(size_t)k * ld];
    unsigned hi = base[(size_t)(k + 1) * ld];
    f.u[p] = lo | (hi << 16);
  }
  return f.v;
}

// ---------------- Tensor Data Mover: 2D tile global -> LDS ----------------
// D# per ISA 08_async_tensor §8.3/8.4. group0: count=1, lds_addr, 57-bit
// global addr, type=2. group1: data_size=2B, tensor_dim0/1, tile_dim0/1,
// tensor_dim0_stride. groups 2/3/4 unused (2D). Issue once per wave; tracked
// with TENSORcnt. 6-arg builtin form (clang-23 / therock headers).
__device__ __forceinline__ void tdm_load_2d(unsigned lds_addr,
                                            const void* gaddr,
                                            unsigned tensor_d0, unsigned tensor_d1,
                                            unsigned tile_d0, unsigned tile_d1,
                                            unsigned stride0_elems) {
  unsigned long long ga = (unsigned long long)(uintptr_t)gaddr;
  tdm_g0_t g0;
  g0[0] = 1u;                                             // count=1, user mode
  g0[1] = lds_addr;                                       // LDS byte address
  g0[2] = (unsigned)(ga & 0xFFFFFFFFu);                   // global_addr[31:0]
  g0[3] = (unsigned)((ga >> 32) & 0x01FFFFFFu) | (2u << 30); // [56:32] | type=2
  tdm_g1_t g1;
  g1[0] = (int)(1u << 16);                                // data_size=1 (2B)
  g1[1] = (int)((tensor_d0 & 0xFFFFu) << 16);             // tensor_dim0 lo16
  g1[2] = (int)((tensor_d0 >> 16) | ((tensor_d1 & 0xFFFFu) << 16));
  g1[3] = (int)((tensor_d1 >> 16) | ((tile_d0 & 0xFFFFu) << 16));
  g1[4] = (int)(tile_d1 & 0xFFFFu);                       // tile_dim1, tile_dim2=0
  g1[5] = (int)stride0_elems;                             // dim0 stride lo32
  g1[6] = 0;
  g1[7] = 0;
  tdm_g2_t z4 = {0, 0, 0, 0};
  tdm_g4_t z8 = {0, 0, 0, 0, 0, 0, 0, 0};
  __builtin_amdgcn_tensor_load_to_lds(g0, g1, z4, z4, z8, 0);
}

// ---------------- elementwise kernels ----------------
__global__ void cvt_f32_bf16(const float* __restrict__ in,
                             unsigned short* __restrict__ out, int n) {
  int i = blockIdx.x * 256 + threadIdx.x;
  int stride = gridDim.x * 256;
  for (; i < n; i += stride) out[i] = f2b(in[i]);
}

__global__ void rmsnorm_kernel(const float* __restrict__ in,
                               const float* __restrict__ w,
                               unsigned short* __restrict__ out,
                               int cols, int in_stride, int out_stride) {
  const int row = blockIdx.x;
  const float* x = in + (size_t)row * in_stride;
  unsigned short* o = out + (size_t)row * out_stride;
  __shared__ float red[256];
  float ss = 0.f;
  for (int c = threadIdx.x; c < cols; c += 256) { float v = x[c]; ss += v * v; }
  red[threadIdx.x] = ss;
  __syncthreads();
  for (int off = 128; off > 0; off >>= 1) {
    if ((int)threadIdx.x < off) red[threadIdx.x] += red[threadIdx.x + off];
    __syncthreads();
  }
  float inv = rsqrtf(red[0] / (float)cols + 1e-6f);
  for (int c = threadIdx.x; c < cols; c += 256) o[c] = f2b(w[c] * x[c] * inv);
}

// rope on q_pe (bf16 q buffer) -> last 64 dims of qfull[h][s][.]
__global__ void rope_q_kernel(const unsigned short* __restrict__ q,
                              const int* __restrict__ pos,
                              unsigned short* __restrict__ qfull) {
  int idx = blockIdx.x * 256 + threadIdx.x;     // S*H*32
  if (idx >= S_ * H_ * 32) return;
  int i = idx & 31;
  int h = (idx >> 5) & (H_ - 1);
  int s = idx >> 10;
  float p = (float)pos[s];
  float inv = powf(10000.0f, -((float)(2 * i) / (float)DR_));
  float f = p * inv;
  float cs = cosf(f), sn = sinf(f);
  const unsigned short* qp = q + (size_t)s * (H_ * QDIM) + h * QDIM + DN_;
  float x1 = b2f(qp[i]), x2 = b2f(qp[i + 32]);
  unsigned short* o = qfull + (size_t)h * S_ * CFULL + (size_t)s * CFULL + KVL_;
  o[i]      = f2b(x1 * cs - x2 * sn);
  o[i + 32] = f2b(x2 * cs + x1 * sn);
}

// rope on k_pe (f32 ckv_pe buffer) -> last 64 dims of kfull[s][.]
__global__ void rope_k_kernel(const float* __restrict__ ckvpe,
                              const int* __restrict__ pos,
                              unsigned short* __restrict__ kfull) {
  int idx = blockIdx.x * 256 + threadIdx.x;     // S*32
  if (idx >= S_ * 32) return;
  int i = idx & 31;
  int s = idx >> 5;
  float p = (float)pos[s];
  float inv = powf(10000.0f, -((float)(2 * i) / (float)DR_));
  float f = p * inv;
  float cs = cosf(f), sn = sinf(f);
  const float* kp = ckvpe + (size_t)s * CFULL + KVL_;
  float x1 = kp[i], x2 = kp[i + 32];
  unsigned short* o = kfull + (size_t)s * CFULL + KVL_;
  o[i]      = f2b(x1 * cs - x2 * sn);
  o[i + 32] = f2b(x2 * cs + x1 * sn);
}

// ---------------- tiled bf16 WMMA GEMM (double-buffered LDS) ----------------
// C[M,N] = A[M,K] * B[K,N], bf16 in, f32 accum, f32 or bf16 out.
// Block tile 128x128x32, 8 wave32 waves (4x2), each wave 2x4 16x16 WMMA tiles.
// Ping-pong LDS tiles: next K-step's global->LDS overlaps current WMMAs.
// Call-site invariants: M%128==0, K%32==0, lda/ldb%8==0, 16B-aligned bases.
__global__ void __launch_bounds__(256)
gemm_bf16_kernel(const unsigned short* __restrict__ A,
                 const unsigned short* __restrict__ B,
                 void* __restrict__ C,
                 int M, int N, int K, int lda, int ldb, int ldc,
                 long aBatch, long bBatch, long cBatch, int outBf16) {
  A += (size_t)blockIdx.z * aBatch;
  B += (size_t)blockIdx.z * bBatch;
  const int tid   = threadIdx.x;
  const int wave  = tid >> 5;
  const int lane  = tid & 31;
  const int l15   = lane & 15;
  const int half8 = (lane & 16) ? 8 : 0;
  const int waveM = wave >> 1;          // 0..3 -> 32 rows each
  const int waveN = wave & 1;           // 0..1 -> 64 cols each
  const int blockM = blockIdx.y * 128;
  const int blockN = blockIdx.x * 128;

  __shared__ __align__(16) unsigned short As[2][128 * 32];  // [m][k]
  __shared__ __align__(16) unsigned short Bs[2][128 * 32];  // transposed [n][k]

  AccF acc[2][4];
#pragma unroll
  for (int mt = 0; mt < 2; ++mt)
#pragma unroll
    for (int nt = 0; nt < 4; ++nt)
#pragma unroll
      for (int r = 0; r < 8; ++r) acc[mt][nt].f[r] = 0.f;

  auto load_tiles = [&](int k0, int bufi) {
    // A tile 128x32 : 512 uint4, 2 per thread (b128 load -> b128 LDS store)
#pragma unroll
    for (int i = 0; i < 2; ++i) {
      int t2  = tid * 2 + i;
      int row = t2 >> 2;
      int c4  = t2 & 3;
      const uint4* src =
          (const uint4*)(A + (size_t)(blockM + row) * lda + k0 + c4 * 8);
      *(uint4*)(&As[bufi][row * 32 + c4 * 8]) = *src;
    }
    // B tile 32x128, transpose into Bs[n][k]
#pragma unroll
    for (int i = 0; i < 2; ++i) {
      int t2  = tid * 2 + i;
      int row = t2 >> 4;                 // 0..31 (k)
      int c4  = t2 & 15;                 // 0..15 (8-col groups)
      int coln = blockN + c4 * 8;
      uint4 v = make_uint4(0u, 0u, 0u, 0u);
      if (coln < N) v = *(const uint4*)(B + (size_t)(k0 + row) * ldb + coln);
      union { uint4 q; unsigned short h[8]; } u; u.q = v;
#pragma unroll
      for (int j = 0; j < 8; ++j) Bs[bufi][(c4 * 8 + j) * 32 + row] = u.h[j];
    }
  };

  load_tiles(0, 0);
  int buf = 0;
  for (int k0 = 0; k0 < K; k0 += 32) {
    __syncthreads();
    if (k0 + 32 < K) load_tiles(k0 + 32, buf ^ 1);

    BFrag a[2], b[4];
#pragma unroll
    for (int mt = 0; mt < 2; ++mt)
      a[mt].v = load_frag_row(&As[buf][(waveM * 32 + mt * 16 + l15) * 32]);
#pragma unroll
    for (int nt = 0; nt < 4; ++nt)
      b[nt].v = load_frag_row(&Bs[buf][(waveN * 64 + nt * 16 + l15) * 32]);
#pragma unroll
    for (int mt = 0; mt < 2; ++mt)
#pragma unroll
      for (int nt = 0; nt < 4; ++nt)
        acc[mt][nt].v = wmma_bf16(a[mt].v, b[nt].v, acc[mt][nt].v);
    buf ^= 1;
  }

  // epilogue: C/D layout per ISA: lane<16 -> M=r, lane>=16 -> M=8+r, N=lane&15
#pragma unroll
  for (int mt = 0; mt < 2; ++mt) {
#pragma unroll
    for (int nt = 0; nt < 4; ++nt) {
      int n = blockN + waveN * 64 + nt * 16 + l15;
      if (n < N) {
#pragma unroll
        for (int r = 0; r < 8; ++r) {
          int m = blockM + waveM * 32 + mt * 16 + half8 + r;
          float v = acc[mt][nt].f[r];
          if (outBf16)
            ((unsigned short*)C)[(size_t)blockIdx.z * cBatch + (size_t)m * ldc + n] = f2b(v);
          else
            ((float*)C)[(size_t)blockIdx.z * cBatch + (size_t)m * ldc + n] = v;
        }
      }
    }
  }
}

// ---------------- flash-style MLA attention ----------------
// One block = 16 queries of one head. 8 waves. Keys processed in chunks of 32.
// K-chunk (32x576 bf16) staged into LDS by the Tensor Data Mover (one
// tensor_load_to_lds per chunk, wave 0 only, TENSORcnt-tracked). Scores by
// waves 0/1 over c=576; online softmax; P*ckv with each wave owning a 64-wide
// slice of the 512-dim value space.
__global__ void __launch_bounds__(256)
flash_mla_kernel(const unsigned short* __restrict__ qfull,
                 const unsigned short* __restrict__ kfull,
                 unsigned short* __restrict__ attn) {
  const int qbase = blockIdx.x * 16;
  const int head  = blockIdx.y;
  const int tid   = threadIdx.x;
  const int wave  = tid >> 5;
  const int lane  = tid & 31;
  const int l15   = lane & 15;
  const int half8 = (lane & 16) ? 8 : 0;

  __shared__ __align__(16) unsigned short qs[16 * CFULL];  // 18 KB
  __shared__ __align__(16) unsigned short ks[32 * CFULL];  // 36 KB
  __shared__ __align__(16) float          sc[16 * 32];
  __shared__ __align__(16) unsigned short pb[16 * 32];
  __shared__ float rowM[16], rowL[16], rowScale[16];

  // q tile: 16 rows x 576 contiguous -> 1152 uint4
  const unsigned short* qsrc =
      qfull + (size_t)head * S_ * CFULL + (size_t)qbase * CFULL;
  for (int i = tid; i < 16 * (CFULL / 8); i += 256)
    ((uint4*)qs)[i] = ((const uint4*)qsrc)[i];

  if (tid < 16) { rowM[tid] = -3.0e38f; rowL[tid] = 0.f; }

  AccF acc[4];
#pragma unroll
  for (int nt = 0; nt < 4; ++nt)
#pragma unroll
    for (int r = 0; r < 8; ++r) acc[nt].f[r] = 0.f;

  const int nChunks = (qbase + 15) / 32 + 1;
  for (int ch = 0; ch < nChunks; ++ch) {
    const int kbase = ch * 32;
    __syncthreads();   // previous PV done (and q/rowM visible on iter 0)
    if (wave == 0) {
      // TDM: 32x576 bf16 tile of kfull -> ks (EXEC ignored; one issue/wave)
      tdm_load_2d((unsigned)(uintptr_t)ks,
                  kfull + (size_t)kbase * CFULL,
                  CFULL, (unsigned)(S_ - kbase),
                  CFULL, 32, CFULL);
      __builtin_amdgcn_s_wait_tensorcnt(0);
    }
    __syncthreads();

    // scores: waves 0,1 each compute a 16x16 tile (keys kbase+16w..)
    if (wave < 2) {
      AccF s;
#pragma unroll
      for (int r = 0; r < 8; ++r) s.f[r] = 0.f;
      for (int kk = 0; kk < CFULL; kk += 32) {
        BFrag a, b;
        a.v = load_frag_row(&qs[l15 * CFULL + kk]);
        b.v = load_frag_row(&ks[(wave * 16 + l15) * CFULL + kk]);
        s.v = wmma_bf16(a.v, b.v, s.v);
      }
#pragma unroll
      for (int r = 0; r < 8; ++r) {
        int m  = half8 + r;
        int kl = wave * 16 + l15;
        float v = s.f[r] * SCALE_;
        if (kbase + kl > qbase + m) v = -3.0e38f;   // causal mask
        sc[m * 32 + kl] = v;
      }
    }
    __syncthreads();

    // online softmax, one thread per query row
    if (tid < 16) {
      float mOld = rowM[tid];
      float mNew = mOld;
      for (int j = 0; j < 32; ++j) mNew = fmaxf(mNew, sc[tid * 32 + j]);
      float scale = expf(mOld - mNew);
      float sum = 0.f;
      for (int j = 0; j < 32; ++j) {
        float p = expf(sc[tid * 32 + j] - mNew);
        sum += p;
        pb[tid * 32 + j] = f2b(p);
      }
      rowL[tid] = rowL[tid] * scale + sum;
      rowScale[tid] = scale;
      rowM[tid] = mNew;
    }
    __syncthreads();

    // rescale accumulators + P*V (each wave: 64 value dims)
    float s8[8];
#pragma unroll
    for (int r = 0; r < 8; ++r) s8[r] = rowScale[half8 + r];
    BFrag a;
    a.v = load_frag_row(&pb[l15 * 32]);
#pragma unroll
    for (int nt = 0; nt < 4; ++nt) {
#pragma unroll
      for (int r = 0; r < 8; ++r) acc[nt].f[r] *= s8[r];
      BFrag b;
      b.v = load_frag_col(&ks[wave * 64 + nt * 16 + l15], CFULL);
      acc[nt].v = wmma_bf16(a.v, b.v, acc[nt].v);
    }
  }
  __syncthreads();

  float l8[8];
#pragma unroll
  for (int r = 0; r < 8; ++r) l8[r] = 1.f / rowL[half8 + r];
#pragma unroll
  for (int nt = 0; nt < 4; ++nt) {
    int d = wave * 64 + nt * 16 + l15;
#pragma unroll
    for (int r = 0; r < 8; ++r) {
      int m = half8 + r;
      attn[(size_t)(qbase + m) * H_ * KVL_ + (size_t)head * KVL_ + d] =
          f2b(acc[nt].f[r] * l8[r]);
    }
  }
}

// ---------------- host orchestration ----------------
extern "C" void kernel_launch(void* const* d_in, const int* in_sizes, int n_in,
                              void* d_out, int out_size, void* d_ws, size_t ws_size,
                              hipStream_t stream) {
  const float* h     = (const float*)d_in[0];
  const int*   pos   = (const int*)d_in[1];
  const float* Wq_a  = (const float*)d_in[2];
  const float* q_ln  = (const float*)d_in[3];
  const float* Wq_b  = (const float*)d_in[4];
  const float* Wkv_a = (const float*)d_in[5];
  const float* kv_ln = (const float*)d_in[6];
  const float* w_kc  = (const float*)d_in[7];
  const float* w_vc  = (const float*)d_in[8];
  const float* Wo    = (const float*)d_in[9];
  float* out = (float*)d_out;

  char* ws = (char*)d_ws;
  size_t off = 0;
  auto alloc = [&](size_t bytes) {
    void* p = ws + off;
    off = (off + bytes + 255) & ~(size_t)255;
    return p;
  };

  unsigned short* hb    = (unsigned short*)alloc((size_t)S_ * HID_ * 2);
  unsigned short* wqa   = (unsigned short*)alloc((size_t)HID_ * QL_ * 2);
  unsigned short* wqb   = (unsigned short*)alloc((size_t)QL_ * H_ * QDIM * 2);
  unsigned short* wkva  = (unsigned short*)alloc((size_t)HID_ * CFULL * 2);
  unsigned short* wkc   = (unsigned short*)alloc((size_t)H_ * DN_ * KVL_ * 2);
  unsigned short* wvc   = (unsigned short*)alloc((size_t)H_ * KVL_ * DV_ * 2);
  unsigned short* wo    = (unsigned short*)alloc((size_t)H_ * DV_ * HID_ * 2);
  float*          qa_f  = (float*)alloc((size_t)S_ * QL_ * 4);
  unsigned short* qan   = (unsigned short*)alloc((size_t)S_ * QL_ * 2);
  unsigned short* q_b   = (unsigned short*)alloc((size_t)S_ * H_ * QDIM * 2);
  float*          ckvpe = (float*)alloc((size_t)S_ * CFULL * 4);
  unsigned short* qfull = (unsigned short*)alloc((size_t)H_ * S_ * CFULL * 2);
  unsigned short* kfull = (unsigned short*)alloc((size_t)S_ * CFULL * 2);
  unsigned short* attn  = (unsigned short*)alloc((size_t)S_ * H_ * KVL_ * 2);
  unsigned short* o_b   = (unsigned short*)alloc((size_t)S_ * H_ * DV_ * 2);

  auto cvt = [&](const float* src, unsigned short* dst, size_t n) {
    cvt_f32_bf16<<<dim3((unsigned)((n + 255) / 256)), dim3(256), 0, stream>>>(
        src, dst, (int)n);
  };
  auto gemm = [&](const unsigned short* A, const unsigned short* B, void* C,
                  int M, int N, int K, int lda, int ldb, int ldc,
                  long aB, long bB, long cB, int batch, int outBf16) {
    dim3 grid((N + 127) / 128, M / 128, batch);
    gemm_bf16_kernel<<<grid, dim3(256), 0, stream>>>(A, B, C, M, N, K, lda, ldb,
                                                     ldc, aB, bB, cB, outBf16);
  };

  // 1) convert activations + weights to bf16
  cvt(h,     hb,   (size_t)S_ * HID_);
  cvt(Wq_a,  wqa,  (size_t)HID_ * QL_);
  cvt(Wq_b,  wqb,  (size_t)QL_ * H_ * QDIM);
  cvt(Wkv_a, wkva, (size_t)HID_ * CFULL);
  cvt(w_kc,  wkc,  (size_t)H_ * DN_ * KVL_);
  cvt(w_vc,  wvc,  (size_t)H_ * KVL_ * DV_);
  cvt(Wo,    wo,   (size_t)H_ * DV_ * HID_);

  // 2) q_a = h @ Wq_a (f32 out for rmsnorm)
  gemm(hb, wqa, qa_f, S_, QL_, HID_, HID_, QL_, QL_, 0, 0, 0, 1, 0);
  // 3) rmsnorm(q_a) -> bf16
  rmsnorm_kernel<<<S_, 256, 0, stream>>>(qa_f, q_ln, qan, QL_, QL_, QL_);
  // 4) q = q_a_n @ Wq_b (bf16 out)
  gemm(qan, wqb, q_b, S_, H_ * QDIM, QL_, QL_, H_ * QDIM, H_ * QDIM,
       0, 0, 0, 1, 1);
  // 5) ckv_pe = h @ Wkv_a (f32 out, N=576 guarded)
  gemm(hb, wkva, ckvpe, S_, CFULL, HID_, HID_, CFULL, CFULL, 0, 0, 0, 1, 0);
  // 6) kfull[:, :512] = rmsnorm(ckv), kfull[:, 512:] = rope(k_pe)
  rmsnorm_kernel<<<S_, 256, 0, stream>>>(ckvpe, kv_ln, kfull, KVL_, CFULL, CFULL);
  rope_k_kernel<<<(S_ * 32 + 255) / 256, 256, 0, stream>>>(ckvpe, pos, kfull);
  // 7) qfull[:, :, 512:] = rope(q_pe)
  rope_q_kernel<<<(S_ * H_ * 32 + 255) / 256, 256, 0, stream>>>(q_b, pos, qfull);
  // 8) qfull[h, :, :512] = q_nope @ w_kc[h]   (batched over heads)
  gemm(q_b, wkc, qfull, S_, KVL_, DN_, H_ * QDIM, KVL_, CFULL,
       (long)QDIM, (long)DN_ * KVL_, (long)S_ * CFULL, H_, 1);
  // 9) attention core (TDM-fed)
  flash_mla_kernel<<<dim3(S_ / 16, H_), dim3(256), 0, stream>>>(qfull, kfull, attn);
  // 10) o[:, h*128:(h+1)*128] = attn[:, h, :] @ w_vc[h]
  gemm(attn, wvc, o_b, S_, DV_, KVL_, H_ * KVL_, DV_, H_ * DV_,
       (long)KVL_, (long)KVL_ * DV_, (long)DV_, H_, 1);
  // 11) out = o @ Wo (f32)
  gemm(o_b, wo, out, S_, HID_, H_ * DV_, H_ * DV_, HID_, HID_, 0, 0, 0, 1, 0);
}
